// Attention_57844619542638
// MI455X (gfx1250) — matrix operand order repaired and checked by reading, not compiled, not run on hardware
//
#include <hip/hip_runtime.h>
#include <hip/hip_bf16.h>

// Shapes from the reference
#define SRC_LEN 2048
#define BS      32
#define ENC_FEAT 1024   // K
#define DEC_HID  1024   // N
#define M_TOTAL (SRC_LEN * BS)   // 65536, m = s*32 + b (enc memory order)

typedef __attribute__((ext_vector_type(16))) __bf16 v16bf;
typedef __attribute__((ext_vector_type(8)))  float  v8f;

union FragU { uint4 q[2]; v16bf v; };

__device__ __forceinline__ unsigned short f2bf(float x) {
    unsigned u = __float_as_uint(x);
    unsigned r = (u + 0x7FFFu + ((u >> 16) & 1u)) >> 16;  // RNE
    return (unsigned short)r;
}

// pack two f32 -> two bf16 (round-half-up) with one v_perm_b32
__device__ __forceinline__ unsigned pkbf(float x, float y) {
    unsigned ux = __float_as_uint(x) + 0x8000u;
    unsigned uy = __float_as_uint(y) + 0x8000u;
    // D bytes[3:2] = uy[3:2], D bytes[1:0] = ux[3:2]
    return __builtin_amdgcn_perm(uy, ux, 0x07060302u);
}

// ---------------------------------------------------------------- init
__global__ void init_kernel(float* __restrict__ scores, float* __restrict__ proj,
                            const float* __restrict__ b_attn) {
    int i = blockIdx.x * 256 + threadIdx.x;   // 98304 threads
    if (i < M_TOTAL) {
        scores[i] = 0.f;
    } else {
        int j = i - M_TOTAL;                  // 0..32767
        proj[j] = b_attn[j & 1023];
    }
}

// ---------------------------------------------------------------- W_e -> bf16
__global__ void convert_w_kernel(const float* __restrict__ W,
                                 unsigned short* __restrict__ Wbf) {
    int g = blockIdx.x * 256 + threadIdx.x;   // 262144 threads, 4 elems each
    int d = g >> 8;
    int e = (g & 255) << 2;
    float4 v = *(const float4*)(W + (size_t)d * 2048 + e);
    uint2 o;
    o.x = (unsigned)f2bf(v.x) | ((unsigned)f2bf(v.y) << 16);
    o.y = (unsigned)f2bf(v.z) | ((unsigned)f2bf(v.w) << 16);
    *(uint2*)(Wbf + (size_t)d * 1024 + e) = o;
}

// ---------------------------------------------------------------- dec_proj
// proj[b,d] += sum_j dec[b,j] * W_attn[d, 1024+j]   (split over j-chunks, atomics)
__global__ void proj_kernel(const float* __restrict__ dec,
                            const float* __restrict__ W,
                            float* __restrict__ proj) {
    __shared__ float sdec[32][256];
    int t = threadIdx.x;
    int j0 = blockIdx.y * 256;
    for (int idx = t; idx < 32 * 256; idx += 256) {
        int b = idx >> 8, jj = idx & 255;
        sdec[b][jj] = dec[b * 1024 + j0 + jj];
    }
    __syncthreads();
    int d = blockIdx.x * 256 + t;
    float acc[32];
#pragma unroll
    for (int b = 0; b < 32; b++) acc[b] = 0.f;
    const float* wrow = W + (size_t)d * 2048 + 1024 + j0;
    for (int jj = 0; jj < 256; jj++) {
        float w = wrow[jj];
#pragma unroll
        for (int b = 0; b < 32; b++) acc[b] += w * sdec[b][jj];
    }
#pragma unroll
    for (int b = 0; b < 32; b++) atomicAdd(&proj[b * 1024 + d], acc[b]);
}

// ---------------------------------------------------------------- fused GEMM
// C[m,d] = sum_e enc[m,e]*W_e[d,e]; score[m] += sum_d wcomb[d]*tanh(C+proj[b,d])
// Block: 128 M x 128 N, 256 threads = 8 waves (2 M x 4 N), wave = 64x32 = 4x2 frags.
// Double-buffered LDS pipeline: A staged via VGPRs (f32->bf16 convert),
// B staged with GLOBAL_LOAD_ASYNC_TO_LDS_B128 (ASYNCcnt).
__global__ void __launch_bounds__(256)
gemm_kernel(const float* __restrict__ enc,
            const unsigned short* __restrict__ Wbf,
            const float* __restrict__ proj,
            const float* __restrict__ wcomb,
            float* __restrict__ scores) {
    __shared__ __align__(16) unsigned short sA[2][128][72];  // 2 x 18KB
    __shared__ __align__(16) unsigned short sB[2][128][72];  // 2 x 18KB
    __shared__ float sProj[32][128];                         // 16KB

    int t = threadIdx.x;
    int lane = t & 31, wave = t >> 5;
    int waveM = wave >> 2, waveN = wave & 3;
    int hf = lane >> 4, lq = lane & 15;
    int bm = blockIdx.y, bn = blockIdx.x;

    // proj slice for this N-tile: all 32 batches x 128 d's
    for (int idx = t; idx < 32 * 128; idx += 256) {
        int b = idx >> 7, dd = idx & 127;
        sProj[b][dd] = proj[b * 1024 + bn * 128 + dd];
    }

    v8f acc[4][2];
#pragma unroll
    for (int i = 0; i < 4; i++)
#pragma unroll
        for (int j = 0; j < 2; j++)
#pragma unroll
            for (int r = 0; r < 8; r++) acc[i][j][r] = 0.f;

    const float* encBase = enc + (size_t)(bm * 128) * 1024;
    const unsigned short* wBase = Wbf + (size_t)(bn * 128) * 1024;

    // per-thread staging geometry
    const int aRow = (t + 0 * 256) >> 4;       // recomputed per i below anyway
    (void)aRow;
    float4 areg[8];

    // issue A-tile global loads (kept in VGPRs until store phase)
#define LOAD_A(KT)                                                             \
    _Pragma("unroll")                                                          \
    for (int i = 0; i < 8; i++) {                                              \
        int idx = t + i * 256;                                                 \
        int row = idx >> 4;                                                    \
        int k4  = (idx & 15) << 2;                                             \
        areg[i] = *(const float4*)(encBase + (size_t)row * 1024 + (KT) * 64 + k4); \
    }

    // convert + store A-tile into LDS buffer (contiguous K order)
#define STORE_A(BUF)                                                           \
    _Pragma("unroll")                                                          \
    for (int i = 0; i < 8; i++) {                                              \
        int idx = t + i * 256;                                                 \
        int row = idx >> 4;                                                    \
        int k4  = (idx & 15) << 2;                                             \
        int kb = k4 & 31, kbase = k4 & 32;                                     \
        int j = k4;                                                            \
        *(uint2*)&sA[BUF][row][j] =                                            \
            make_uint2(pkbf(areg[i].x, areg[i].y), pkbf(areg[i].z, areg[i].w)); \
    }

    // async global->LDS copy of B-tile (bf16 weights), tracked by ASYNCcnt
#define LOAD_B_ASYNC(KT, BUF)                                                  \
    _Pragma("unroll")                                                          \
    for (int i = 0; i < 4; i++) {                                              \
        int idx = t + i * 256;                                                 \
        int row = idx >> 3;                                                    \
        int e8  = (idx & 7) << 3;                                              \
        unsigned ldsOff = (unsigned)(size_t)&sB[BUF][row][e8];                 \
        unsigned gOff = ((unsigned)row * 1024u + (unsigned)((KT) * 64 + e8)) * 2u; \
        asm volatile("global_load_async_to_lds_b128 %0, %1, %2"                \
                     :: "v"(ldsOff), "v"(gOff), "s"(wBase) : "memory");        \
    }

#define WAIT_ASYNC() asm volatile("s_wait_asynccnt 0x0" ::: "memory")

    // ---- prologue: fill buffer 0
    LOAD_A(0);
    LOAD_B_ASYNC(0, 0);
    STORE_A(0);            // compiler inserts s_wait_loadcnt before converts
    WAIT_ASYNC();
    __syncthreads();

    for (int kt = 0; kt < 16; ++kt) {        // K = 16 * 64
        int cur = kt & 1;
        int nxt = cur ^ 1;

        // issue next tile's global traffic before computing (latency overlap)
        if (kt < 15) {
            LOAD_A(kt + 1);
            LOAD_B_ASYNC(kt + 1, nxt);
        }

        const unsigned short (*A)[72] = sA[cur];
        const unsigned short (*B)[72] = sB[cur];
#pragma unroll
        for (int ks = 0; ks < 64; ks += 32) {
            FragU a[4], bb[2];
#pragma unroll
            for (int fm = 0; fm < 4; fm++) {
                const uint4* p = (const uint4*)&A[waveM * 64 + fm * 16 + lq][ks + hf * 16];
                a[fm].q[0] = p[0]; a[fm].q[1] = p[1];
            }
#pragma unroll
            for (int fn = 0; fn < 2; fn++) {
                const uint4* p = (const uint4*)&B[waveN * 32 + fn * 16 + lq][ks + hf * 16];
                bb[fn].q[0] = p[0]; bb[fn].q[1] = p[1];
            }
#pragma unroll
            for (int fm = 0; fm < 4; fm++)
#pragma unroll
                for (int fn = 0; fn < 2; fn++)
                    acc[fm][fn] = __builtin_amdgcn_wmma_f32_16x16x32_bf16(
                        false, a[fm].v, false, bb[fn].v,
                        (short)0, acc[fm][fn], false, false);
        }

        if (kt < 15) {
            STORE_A(nxt);      // waits loadcnt, converts, ds_store
            WAIT_ASYNC();      // B tile landed in LDS
            __syncthreads();   // publish buffer nxt
        }
    }

    // epilogue: tanh + w_comb dot, 16-lane butterfly, atomic partial scores
    float wv[2];
#pragma unroll
    for (int fn = 0; fn < 2; fn++)
        wv[fn] = wcomb[bn * 128 + waveN * 32 + fn * 16 + lq];

#pragma unroll
    for (int fm = 0; fm < 4; fm++) {
#pragma unroll
        for (int r = 0; r < 8; r++) {
            int mLoc = waveM * 64 + fm * 16 + hf * 8 + r;   // C layout: row = r + 8*(lane/16)
            int mG = bm * 128 + mLoc;
            int b = mG & 31;                                 // m = s*32 + b
            float s = 0.f;
#pragma unroll
            for (int fn = 0; fn < 2; fn++) {
                int dLoc = waveN * 32 + fn * 16 + lq;        // C layout: col = lane%16
                float v = acc[fm][fn][r] + sProj[b][dLoc];
                s += tanhf(v) * wv[fn];
            }
#pragma unroll
            for (int off = 1; off < 16; off <<= 1) s += __shfl_xor(s, off, 32);
            if (lq == 0) atomicAdd(&scores[mG], s);
        }
    }
}

// ---------------------------------------------------------------- softmax over s
__global__ void softmax_kernel(const float* __restrict__ scores, float* __restrict__ out) {
    __shared__ float red[256];
    int b = blockIdx.x, t = threadIdx.x;
    float x[8];
    float mx = -1e30f;
#pragma unroll
    for (int i = 0; i < 8; i++) {
        int s = t + i * 256;
        x[i] = scores[s * 32 + b];
        mx = fmaxf(mx, x[i]);
    }
    red[t] = mx; __syncthreads();
    for (int off = 128; off > 0; off >>= 1) {
        if (t < off) red[t] = fmaxf(red[t], red[t + off]);
        __syncthreads();
    }
    mx = red[0]; __syncthreads();
    float sum = 0.f;
#pragma unroll
    for (int i = 0; i < 8; i++) { x[i] = expf(x[i] - mx); sum += x[i]; }
    red[t] = sum; __syncthreads();
    for (int off = 128; off > 0; off >>= 1) {
        if (t < off) red[t] += red[t + off];
        __syncthreads();
    }
    float inv = 1.f / red[0];
#pragma unroll
    for (int i = 0; i < 8; i++) out[b * 2048 + t + i * 256] = x[i] * inv;
}

// ---------------------------------------------------------------- launch
extern "C" void kernel_launch(void* const* d_in, const int* in_sizes, int n_in,
                              void* d_out, int out_size, void* d_ws, size_t ws_size,
                              hipStream_t stream) {
    const float* dec  = (const float*)d_in[0];   // (32,1024)
    const float* enc  = (const float*)d_in[1];   // (2048,32,1024) -> (m,e)
    const float* Wat  = (const float*)d_in[2];   // (1024,2048)
    const float* batt = (const float*)d_in[3];   // (1024)
    const float* wcmb = (const float*)d_in[4];   // (1024)
    float* out = (float*)d_out;                  // (32,2048)

    char* ws = (char*)d_ws;
    unsigned short* Wbf = (unsigned short*)ws;                        // 2 MB
    float* proj   = (float*)(ws + (size_t)2 * 1024 * 1024);           // 128 KB
    float* scores = (float*)(ws + (size_t)2 * 1024 * 1024 + 128 * 1024); // 256 KB

    init_kernel     <<<384, 256, 0, stream>>>(scores, proj, batt);
    convert_w_kernel<<<1024, 256, 0, stream>>>(Wat, Wbf);
    proj_kernel     <<<dim3(4, 4), 256, 0, stream>>>(dec, Wat, proj);
    gemm_kernel     <<<dim3(8, 512), 256, 0, stream>>>(enc, Wbf, proj, wcmb, scores);
    softmax_kernel  <<<32, 256, 0, stream>>>(scores, out);
}
